// MultiHeadAttention_50165218017969
// MI455X (gfx1250) — compile-verified, matrix operand order checked by reading, not run on hardware
//
#include <hip/hip_runtime.h>
#include <hip/hip_bf16.h>
#include <cstdint>
#include <cstddef>

typedef __bf16 bf16_t;
typedef __attribute__((ext_vector_type(16))) __bf16 v16bf;
typedef __attribute__((ext_vector_type(8)))  __bf16 v8bf;
typedef __attribute__((ext_vector_type(8)))  float  v8f;

#define EMBED 1024
#define HEADS 16
#define HDIM  64
#define BATCH 4
#define SEQ   2048

// ---------------------------------------------------------------------------
// WMMA helpers (CDNA5 / gfx1250, wave32)
// ---------------------------------------------------------------------------
static __device__ __forceinline__ v8f wmma_bf16(v16bf a, v16bf b, v8f c) {
  // D = A(16x32) * B(32x16) + C(16x16), fp32 accumulate
  return __builtin_amdgcn_wmma_f32_16x16x32_bf16(
      /*neg_a=*/false, a, /*neg_b=*/false, b,
      /*c_mod=*/(short)0, c, /*reuse_a=*/false, /*reuse_b=*/false);
}

// Load a 16x32 bf16 operand fragment (A layout; B uses the same pattern when
// the operand is stored [n][k] row-major). Per ISA 16-bit A layout:
//   lanes 0-15  : row = lane,    VGPR0-3 hold k=0..7,  VGPR4-7 hold k=16..23
//   lanes 16-31 : row = lane-16, VGPR0-3 hold k=8..15, VGPR4-7 hold k=24..31
// => two contiguous 16-byte chunks per lane at k = hi*8 and k = 16 + hi*8.
static __device__ __forceinline__ v16bf load_frag(const bf16_t* __restrict__ base,
                                                  int ld) {
  const int lane = threadIdx.x & 31;
  const int r    = lane & 15;
  const int hi   = lane >> 4;
  const bf16_t* p = base + (size_t)r * ld + hi * 8;
  v8bf a = *(const v8bf*)(p);
  v8bf b = *(const v8bf*)(p + 16);
  return __builtin_shufflevector(a, b, 0, 1, 2, 3, 4, 5, 6, 7,
                                        8, 9, 10, 11, 12, 13, 14, 15);
}

// ---------------------------------------------------------------------------
// Kernel 1: fused QKV projection.  out = X @ W^T + b   (M=B*S, N=E, K=E)
// blockIdx.z selects projection (0=q,1=k,2=v).  Writes bf16:
//   q,k -> [B,H,S,64] row-major ; v -> transposed [B,H,64,S]
// ---------------------------------------------------------------------------
__global__ __launch_bounds__(128) void qkv_proj_kernel(
    const float* __restrict__ xq, const float* __restrict__ xk,
    const float* __restrict__ xv,
    const float* __restrict__ Wq, const float* __restrict__ bq,
    const float* __restrict__ Wk, const float* __restrict__ bk,
    const float* __restrict__ Wv, const float* __restrict__ bv,
    bf16_t* __restrict__ Qb, bf16_t* __restrict__ Kb, bf16_t* __restrict__ Vt) {
  const int which = blockIdx.z;
  const float* X    = (which == 0) ? xq : (which == 1) ? xk : xv;
  const float* W    = (which == 0) ? Wq : (which == 1) ? Wk : Wv;
  const float* bias = (which == 0) ? bq : (which == 1) ? bk : bv;

  const int M0 = blockIdx.x * 64;
  const int N0 = blockIdx.y * 64;

  // K-stride padded to 40 elems (80 B): 16-B aligned, conflict-free b128 reads
  __shared__ bf16_t As[64 * 40];
  __shared__ bf16_t Bs[64 * 40];

  const int t     = threadIdx.x;
  const int srow  = t >> 1;      // staging row 0..63
  const int shalf = t & 1;       // staging half (16 floats each)
  const int wave  = t >> 5;
  const int wm    = wave >> 1;   // wave m-offset (0/1 -> 0/32)
  const int wn    = wave & 1;    // wave n-offset (0/1 -> 0/32)
  const int lane  = t & 31;
  const int ln    = lane & 15;
  const int hi    = lane >> 4;

  v8f acc[4] = {};  // [mt*2+nt]

  for (int k0 = 0; k0 < EMBED; k0 += 32) {
    // ---- stage A (activations) and B (weight rows) as bf16 into LDS ----
    {
      const float* xp = X + (size_t)(M0 + srow) * EMBED + k0 + shalf * 16;
      float4 f0 = ((const float4*)xp)[0];
      float4 f1 = ((const float4*)xp)[1];
      float4 f2 = ((const float4*)xp)[2];
      float4 f3 = ((const float4*)xp)[3];
      v8bf c0 = {(bf16_t)f0.x, (bf16_t)f0.y, (bf16_t)f0.z, (bf16_t)f0.w,
                 (bf16_t)f1.x, (bf16_t)f1.y, (bf16_t)f1.z, (bf16_t)f1.w};
      v8bf c1 = {(bf16_t)f2.x, (bf16_t)f2.y, (bf16_t)f2.z, (bf16_t)f2.w,
                 (bf16_t)f3.x, (bf16_t)f3.y, (bf16_t)f3.z, (bf16_t)f3.w};
      *(v8bf*)&As[srow * 40 + shalf * 16]     = c0;
      *(v8bf*)&As[srow * 40 + shalf * 16 + 8] = c1;

      const float* wp = W + (size_t)(N0 + srow) * EMBED + k0 + shalf * 16;
      float4 g0 = ((const float4*)wp)[0];
      float4 g1 = ((const float4*)wp)[1];
      float4 g2 = ((const float4*)wp)[2];
      float4 g3 = ((const float4*)wp)[3];
      v8bf d0 = {(bf16_t)g0.x, (bf16_t)g0.y, (bf16_t)g0.z, (bf16_t)g0.w,
                 (bf16_t)g1.x, (bf16_t)g1.y, (bf16_t)g1.z, (bf16_t)g1.w};
      v8bf d1 = {(bf16_t)g2.x, (bf16_t)g2.y, (bf16_t)g2.z, (bf16_t)g2.w,
                 (bf16_t)g3.x, (bf16_t)g3.y, (bf16_t)g3.z, (bf16_t)g3.w};
      *(v8bf*)&Bs[srow * 40 + shalf * 16]     = d0;
      *(v8bf*)&Bs[srow * 40 + shalf * 16 + 8] = d1;
    }
    __syncthreads();

    v16bf a0 = load_frag(&As[(wm * 32 + 0) * 40], 40);
    v16bf a1 = load_frag(&As[(wm * 32 + 16) * 40], 40);
    v16bf b0 = load_frag(&Bs[(wn * 32 + 0) * 40], 40);
    v16bf b1 = load_frag(&Bs[(wn * 32 + 16) * 40], 40);
    acc[0] = wmma_bf16(a0, b0, acc[0]);
    acc[1] = wmma_bf16(a0, b1, acc[1]);
    acc[2] = wmma_bf16(a1, b0, acc[2]);
    acc[3] = wmma_bf16(a1, b1, acc[3]);

    __syncthreads();
  }

  // ---- epilogue: add bias, convert to bf16, scatter into head layout ----
#pragma unroll
  for (int nt = 0; nt < 2; ++nt) {
    const int col = N0 + wn * 32 + nt * 16 + ln;
    const float bias_v = bias[col];
    const int h = col >> 6;
    const int d = col & (HDIM - 1);
#pragma unroll
    for (int mt = 0; mt < 2; ++mt) {
      v8f a = acc[mt * 2 + nt];
#pragma unroll
      for (int e = 0; e < 8; ++e) {
        const int row = M0 + wm * 32 + mt * 16 + e + 8 * hi;
        const int b = row >> 11;            // row / SEQ
        const int s = row & (SEQ - 1);      // row % SEQ
        const float val = a[e] + bias_v;
        if (which == 2) {
          Vt[((size_t)(b * HEADS + h) * HDIM + d) * SEQ + s] = (bf16_t)val;
        } else {
          bf16_t* dst = (which == 0) ? Qb : Kb;
          dst[((size_t)(b * HEADS + h) * SEQ + s) * HDIM + d] = (bf16_t)val;
        }
      }
    }
  }
}

// ---------------------------------------------------------------------------
// Kernel 2: flash attention per (b,h).  Wave = 16 queries x Dh=64.
// scores = Q K^T / 8, masked (mask==true -> -1e9), online softmax, O += P V.
// ---------------------------------------------------------------------------
__global__ __launch_bounds__(128) void flash_attn_kernel(
    const bf16_t* __restrict__ Qb, const bf16_t* __restrict__ Kb,
    const bf16_t* __restrict__ Vt, const unsigned char* __restrict__ mask,
    bf16_t* __restrict__ Ao) {
  const int b    = blockIdx.z;
  const int h    = blockIdx.y;
  const int wave = threadIdx.x >> 5;
  const int lane = threadIdx.x & 31;
  const int ln   = lane & 15;
  const int hi   = lane >> 4;
  const int qbase = blockIdx.x * 64 + wave * 16;

  const size_t bh = (size_t)b * HEADS + h;
  const bf16_t* Qp = Qb + bh * SEQ * HDIM;
  const bf16_t* Kp = Kb + bh * SEQ * HDIM;
  const bf16_t* Vp = Vt + bh * HDIM * SEQ;   // transposed: [d][s]
  const unsigned char* mp = mask + (size_t)b * SEQ * SEQ;

  // Per-wave P strip 16x64 bf16, stride 72 elems (144 B, 16-B aligned)
  __shared__ bf16_t Ps[4 * 16 * 72];
  bf16_t* pw = &Ps[wave * 16 * 72];

  // Q fragments (A operand) loaded once, directly from global bf16
  v16bf qf0 = load_frag(Qp + (size_t)qbase * HDIM + 0, HDIM);
  v16bf qf1 = load_frag(Qp + (size_t)qbase * HDIM + 32, HDIM);

  v8f o[4] = {};
  float mrow[8], lrow[8];
#pragma unroll
  for (int e = 0; e < 8; ++e) { mrow[e] = -INFINITY; lrow[e] = 0.0f; }

  for (int kb = 0; kb < SEQ; kb += 64) {
    if (kb + 64 < SEQ) {  // prefetch next K/V chunk into cache
      __builtin_prefetch(Kp + (size_t)(kb + 64) * HDIM + lane * 8, 0, 1);
      __builtin_prefetch(Vp + (size_t)ln * SEQ + kb + 64, 0, 1);
    }

    // ---- scores: 4 tiles of 16 queries x 16 keys ----
    v8f c[4];
#pragma unroll
    for (int j = 0; j < 4; ++j) {
      v16bf k0 = load_frag(Kp + (size_t)(kb + j * 16) * HDIM + 0,  HDIM);
      v16bf k1 = load_frag(Kp + (size_t)(kb + j * 16) * HDIM + 32, HDIM);
      v8f cc = {};
      cc = wmma_bf16(qf0, k0, cc);
      cc = wmma_bf16(qf1, k1, cc);
      const int col = kb + j * 16 + ln;
#pragma unroll
      for (int e = 0; e < 8; ++e) {
        const int row = qbase + e + 8 * hi;
        const unsigned char mm = mp[(size_t)row * SEQ + col];
        cc[e] = mm ? -1.0e9f : cc[e] * 0.125f;
      }
      c[j] = cc;
    }

    // ---- online softmax (row stats live per lane-half, rows r + 8*hi) ----
#pragma unroll
    for (int e = 0; e < 8; ++e) {
      float rmax = fmaxf(fmaxf(c[0][e], c[1][e]), fmaxf(c[2][e], c[3][e]));
#pragma unroll
      for (int off = 1; off < 16; off <<= 1)
        rmax = fmaxf(rmax, __shfl_xor(rmax, off, 32));
      const float mnew  = fmaxf(mrow[e], rmax);
      const float rescale = __expf(mrow[e] - mnew);
      mrow[e] = mnew;
      lrow[e] *= rescale;
      o[0][e] *= rescale; o[1][e] *= rescale;
      o[2][e] *= rescale; o[3][e] *= rescale;
      float psum = 0.0f;
#pragma unroll
      for (int j = 0; j < 4; ++j) {
        const float p = __expf(c[j][e] - mnew);
        c[j][e] = p;
        psum += p;
      }
#pragma unroll
      for (int off = 1; off < 16; off <<= 1)
        psum += __shfl_xor(psum, off, 32);
      lrow[e] += psum;
    }

    // ---- transpose P through LDS (C-layout -> A-operand layout) ----
#pragma unroll
    for (int j = 0; j < 4; ++j)
#pragma unroll
      for (int e = 0; e < 8; ++e)
        pw[(size_t)(e + 8 * hi) * 72 + j * 16 + ln] = (bf16_t)c[j][e];
    asm volatile("s_wait_dscnt 0" ::: "memory");

    v16bf pa0 = load_frag(pw + 0, 72);
    v16bf pa1 = load_frag(pw + 32, 72);

    // ---- O += P (16x64) x V (64x64): B rows = d, contiguous over keys ----
#pragma unroll
    for (int jd = 0; jd < 4; ++jd) {
      v16bf v0 = load_frag(Vp + (size_t)(jd * 16) * SEQ + kb + 0, SEQ);
      v16bf v1 = load_frag(Vp + (size_t)(jd * 16) * SEQ + kb + 32, SEQ);
      o[jd] = wmma_bf16(pa0, v0, o[jd]);
      o[jd] = wmma_bf16(pa1, v1, o[jd]);
    }
  }

  // ---- epilogue: normalize and store bf16 merged-head activations ----
  bf16_t* outp = Ao + (size_t)b * SEQ * EMBED + (size_t)h * HDIM;
#pragma unroll
  for (int e = 0; e < 8; ++e) {
    const float inv = 1.0f / lrow[e];
    const size_t row = (size_t)(qbase + e + 8 * hi);
#pragma unroll
    for (int jd = 0; jd < 4; ++jd)
      outp[row * EMBED + jd * 16 + ln] = (bf16_t)(o[jd][e] * inv);
  }
}

// ---------------------------------------------------------------------------
// Kernel 3: output projection.  out = Ain @ Wo^T + bo  (fp32 result)
// A operand read directly from global bf16; Wo converted through LDS.
// ---------------------------------------------------------------------------
__global__ __launch_bounds__(128) void out_proj_kernel(
    const bf16_t* __restrict__ Ain, const float* __restrict__ Wo,
    const float* __restrict__ bo, float* __restrict__ out) {
  const int M0 = blockIdx.x * 64;
  const int N0 = blockIdx.y * 64;

  __shared__ bf16_t Bs[64 * 40];

  const int t     = threadIdx.x;
  const int srow  = t >> 1;
  const int shalf = t & 1;
  const int wave  = t >> 5;
  const int wm    = wave >> 1;
  const int wn    = wave & 1;
  const int lane  = t & 31;
  const int ln    = lane & 15;
  const int hi    = lane >> 4;

  v8f acc[4] = {};

  for (int k0 = 0; k0 < EMBED; k0 += 32) {
    {
      const float* wp = Wo + (size_t)(N0 + srow) * EMBED + k0 + shalf * 16;
      float4 g0 = ((const float4*)wp)[0];
      float4 g1 = ((const float4*)wp)[1];
      float4 g2 = ((const float4*)wp)[2];
      float4 g3 = ((const float4*)wp)[3];
      v8bf d0 = {(bf16_t)g0.x, (bf16_t)g0.y, (bf16_t)g0.z, (bf16_t)g0.w,
                 (bf16_t)g1.x, (bf16_t)g1.y, (bf16_t)g1.z, (bf16_t)g1.w};
      v8bf d1 = {(bf16_t)g2.x, (bf16_t)g2.y, (bf16_t)g2.z, (bf16_t)g2.w,
                 (bf16_t)g3.x, (bf16_t)g3.y, (bf16_t)g3.z, (bf16_t)g3.w};
      *(v8bf*)&Bs[srow * 40 + shalf * 16]     = d0;
      *(v8bf*)&Bs[srow * 40 + shalf * 16 + 8] = d1;
    }
    __syncthreads();

    v16bf a0 = load_frag(Ain + (size_t)(M0 + wm * 32 + 0) * EMBED + k0, EMBED);
    v16bf a1 = load_frag(Ain + (size_t)(M0 + wm * 32 + 16) * EMBED + k0, EMBED);
    v16bf b0 = load_frag(&Bs[(wn * 32 + 0) * 40], 40);
    v16bf b1 = load_frag(&Bs[(wn * 32 + 16) * 40], 40);
    acc[0] = wmma_bf16(a0, b0, acc[0]);
    acc[1] = wmma_bf16(a0, b1, acc[1]);
    acc[2] = wmma_bf16(a1, b0, acc[2]);
    acc[3] = wmma_bf16(a1, b1, acc[3]);

    __syncthreads();
  }

#pragma unroll
  for (int nt = 0; nt < 2; ++nt) {
    const int col = N0 + wn * 32 + nt * 16 + ln;
    const float bias_v = bo[col];
#pragma unroll
    for (int mt = 0; mt < 2; ++mt) {
      v8f a = acc[mt * 2 + nt];
#pragma unroll
      for (int e = 0; e < 8; ++e) {
        const int row = M0 + wm * 32 + mt * 16 + e + 8 * hi;
        out[(size_t)row * EMBED + col] = a[e] + bias_v;
      }
    }
  }
}

// ---------------------------------------------------------------------------
// Host launcher
// ---------------------------------------------------------------------------
extern "C" void kernel_launch(void* const* d_in, const int* in_sizes, int n_in,
                              void* d_out, int out_size, void* d_ws,
                              size_t ws_size, hipStream_t stream) {
  (void)in_sizes; (void)n_in; (void)out_size; (void)ws_size;

  const float* q  = (const float*)d_in[0];
  const float* k  = (const float*)d_in[1];
  const float* v  = (const float*)d_in[2];
  const unsigned char* mask = (const unsigned char*)d_in[3];  // jax bool: 1B
  const float* Wq = (const float*)d_in[4];
  const float* bq = (const float*)d_in[5];
  const float* Wk = (const float*)d_in[6];
  const float* bk = (const float*)d_in[7];
  const float* Wv = (const float*)d_in[8];
  const float* bv = (const float*)d_in[9];
  const float* Wo = (const float*)d_in[10];
  const float* bo = (const float*)d_in[11];

  char* ws = (char*)d_ws;
  const size_t qkv_bytes = (size_t)BATCH * HEADS * SEQ * HDIM * sizeof(bf16_t);
  bf16_t* Qb = (bf16_t*)(ws + 0 * qkv_bytes);
  bf16_t* Kb = (bf16_t*)(ws + 1 * qkv_bytes);
  bf16_t* Vt = (bf16_t*)(ws + 2 * qkv_bytes);
  bf16_t* Ao = (bf16_t*)(ws + 3 * qkv_bytes);

  dim3 g1(BATCH * SEQ / 64, EMBED / 64, 3);
  qkv_proj_kernel<<<g1, 128, 0, stream>>>(q, k, v, Wq, bq, Wk, bk, Wv, bv,
                                          Qb, Kb, Vt);

  dim3 g2(SEQ / 64, HEADS, BATCH);
  flash_attn_kernel<<<g2, 128, 0, stream>>>(Qb, Kb, Vt, mask, Ao);

  dim3 g3(BATCH * SEQ / 64, EMBED / 64, 1);
  out_proj_kernel<<<g3, 128, 0, stream>>>(Ao, Wo, bo, (float*)d_out);
}
